// HierarchicalFusionSNN_3470333575952
// MI455X (gfx1250) — matrix operand order repaired and checked by reading, not compiled, not run on hardware
//
#include <hip/hip_runtime.h>
#include <hip/hip_bf16.h>
#include <math.h>

// ---------------------------------------------------------------------------
// HierarchicalFusionSNN forward, CDNA5 (gfx1250) WMMA implementation.
//
// All heavy math is f16 WMMA (v_wmma_f32_16x16x32_f16) with f32 accumulate.
// f32 tensors are converted to f16 while staged into LDS. Software-pipelined:
// next tile's global_load_b128 batch is issued before the current tile's
// ds_load/WMMA block; LDS is double buffered -> 1 barrier per 16 WMMAs.
//
// Exact simplifications:
//  * levels 1/2 and global attention have one token -> softmax == 1,
//    so attention(x) == (x @ Wv) @ Wo (temperature/heads drop out).
//  * level 0 attention has 3 tokens -> tiny per-sample softmax kernel.
//
// d_in layout (jax pytree flatten: top-level insertion order audio, vision,
// imu, params; dict keys sorted inside params):
//   0 audio  1 vision  2 imu
//   3..6   gattn Wk,Wo,Wq,Wv
//   7,8    in_proj audio W,b   9,10 in_proj imu W,b   11,12 in_proj vision W,b
//   13..16 L0 attn Wk,Wo,Wq,Wv  17,18 L0 outp W,b
//   19,20  L0 proc audio  21,22 L0 proc imu  23,24 L0 proc vision
//   25..32 L1 (attn x4, outp W,b, proc W,b)   33..40 L2 (same)
//   41,42  readout W(512x1000), b
// Constraints relied upon: M % 128 == 0, K % 64 == 0, N % 4 == 0.
// ---------------------------------------------------------------------------

typedef _Float16 h8   __attribute__((ext_vector_type(8)));
typedef _Float16 v16h __attribute__((ext_vector_type(16)));
typedef float    v8f  __attribute__((ext_vector_type(8)));

#define HDIM 512
#define BK   64
#define KP2  72      // LDS row stride in halfs (64 data + 8 pad; 144B, 16B-aligned)

#define MODE_BIAS 1
#define MODE_LIF  2
#define MODE_RES  4

// ---- staging helpers -------------------------------------------------------
__device__ __forceinline__ void stage_global_A(const float* __restrict__ A, int lda,
                                               int rowBase, int k0, int tid, float4* ra)
{
    const int c4 = tid & 15;          // float4 column 0..15 (64 floats)
    const int r0 = tid >> 4;          // 0..15
    const float* base = A + (size_t)(rowBase + r0) * lda + k0 + c4 * 4;
#pragma unroll
    for (int rr = 0; rr < 8; ++rr)
        ra[rr] = *(const float4*)(base + (size_t)(rr * 16) * lda);
}

__device__ __forceinline__ void stage_global_B(const float* __restrict__ W, int ldb,
                                               int colBase, int k0, int N, int tid, float4* rb)
{
    const int c4  = tid & 31;         // float4 column 0..31 (128 floats)
    const int kk0 = tid >> 5;         // 0..7
    const int gn  = colBase + c4 * 4;
    const int gnc = (gn + 3 < N) ? gn : (N - 4);   // clamp (N%4==0 -> clamped == all-OOB)
    const float* base = W + (size_t)(k0 + kk0) * ldb + gnc;
#pragma unroll
    for (int ki = 0; ki < 8; ++ki)
        rb[ki] = *(const float4*)(base + (size_t)(ki * 8) * ldb);
}

__device__ __forceinline__ void stage_store(_Float16* __restrict__ As, _Float16* __restrict__ Bs,
                                            const float4* ra, const float4* rb,
                                            int tid, bool bok)
{
    {   // A: [row][k], row-major
        const int c4 = tid & 15, r0 = tid >> 4;
#pragma unroll
        for (int rr = 0; rr < 8; ++rr) {
            _Float16* d = As + (r0 + rr * 16) * KP2 + c4 * 4;
            d[0] = (_Float16)ra[rr].x; d[1] = (_Float16)ra[rr].y;
            d[2] = (_Float16)ra[rr].z; d[3] = (_Float16)ra[rr].w;
        }
    }
    {   // B: transposed into [col][k]
        const int c4 = tid & 31, kk0 = tid >> 5;
#pragma unroll
        for (int ki = 0; ki < 8; ++ki) {
            const int kk = kk0 + ki * 8;
            float x0 = bok ? rb[ki].x : 0.f;
            float x1 = bok ? rb[ki].y : 0.f;
            float x2 = bok ? rb[ki].z : 0.f;
            float x3 = bok ? rb[ki].w : 0.f;
            Bs[(c4 * 4 + 0) * KP2 + kk] = (_Float16)x0;
            Bs[(c4 * 4 + 1) * KP2 + kk] = (_Float16)x1;
            Bs[(c4 * 4 + 2) * KP2 + kk] = (_Float16)x2;
            Bs[(c4 * 4 + 3) * KP2 + kk] = (_Float16)x3;
        }
    }
}

// ---------------------------------------------------------------------------
// Generic GEMM: C[M,N] = epilogue(A[M,K] @ W[K,N]); 256 threads = 8 wave32,
// block tile 128x128, K-step 64; wave (2x4 grid) owns 64x32 = 4x2 WMMA tiles.
// ---------------------------------------------------------------------------
template <int MODE>
__global__ __launch_bounds__(256)
void gemm_wmma(const float* __restrict__ A, int lda,
               const float* __restrict__ W, int ldb,
               const float* __restrict__ bias,
               const float* __restrict__ res, int ldr,
               float* __restrict__ C, int ldc,
               int M, int N, int K)
{
    __shared__ _Float16 sA[2][128 * KP2];
    __shared__ _Float16 sB[2][128 * KP2];

    const int tid  = threadIdx.x;
    const int lane = tid & 31;
    const int lane_lo = lane & 15;
    const int sel  = lane >> 4;
    const int w    = tid >> 5;
    const int wm   = w & 1;
    const int wn   = w >> 1;

    const int rowBase = blockIdx.y * 128;
    const int colBase = blockIdx.x * 128;
    const bool bok = (colBase + (tid & 31) * 4 + 3) < N;

    v8f acc[4][2];
#pragma unroll
    for (int i = 0; i < 4; ++i)
#pragma unroll
        for (int j = 0; j < 2; ++j)
            acc[i][j] = (v8f){0.f,0.f,0.f,0.f,0.f,0.f,0.f,0.f};

    float4 ra[8], rb[8];
    const int nK = K / BK;

    // prologue: stage tile 0
    stage_global_A(A, lda, rowBase, 0, tid, ra);
    stage_global_B(W, ldb, colBase, 0, N, tid, rb);
    stage_store(sA[0], sB[0], ra, rb, tid, bok);
    __syncthreads();

    for (int kt = 0; kt < nK; ++kt) {
        const bool has_next = (kt + 1) < nK;
        if (has_next) {           // issue next tile's global loads early
            stage_global_A(A, lda, rowBase, (kt + 1) * BK, tid, ra);
            stage_global_B(W, ldb, colBase, (kt + 1) * BK, N, tid, rb);
        }

        const _Float16* As = sA[kt & 1];
        const _Float16* Bs = sB[kt & 1];
#pragma unroll
        for (int h = 0; h < 2; ++h) {       // two K=32 halves of the BK=64 stage
            v16h af[4], bf[2];
#pragma unroll
            for (int i = 0; i < 4; ++i) {
                const int r = wm * 64 + i * 16 + lane_lo;
                const _Float16* p = As + r * KP2 + h * 32 + sel * 16;
                h8 lo = *(const h8*)p;
                h8 hi = *(const h8*)(p + 8);
                af[i] = __builtin_shufflevector(lo, hi,
                         0,1,2,3,4,5,6,7,8,9,10,11,12,13,14,15);
            }
#pragma unroll
            for (int j = 0; j < 2; ++j) {
                const int c = wn * 32 + j * 16 + lane_lo;
                const _Float16* p = Bs + c * KP2 + h * 32 + sel * 16;
                h8 lo = *(const h8*)p;
                h8 hi = *(const h8*)(p + 8);
                bf[j] = __builtin_shufflevector(lo, hi,
                         0,1,2,3,4,5,6,7,8,9,10,11,12,13,14,15);
            }
#pragma unroll
            for (int i = 0; i < 4; ++i)
#pragma unroll
                for (int j = 0; j < 2; ++j)
                    acc[i][j] = __builtin_amdgcn_wmma_f32_16x16x32_f16(
                        false, af[i], false, bf[j], (short)0, acc[i][j], false, false);
        }

        if (has_next)   // write next tile into the other LDS buffer
            stage_store(sA[(kt + 1) & 1], sB[(kt + 1) & 1], ra, rb, tid, bok);
        __syncthreads();
    }

    // ---- epilogue: bias / LIF spike (x>1 ? 1 : 0) / residual add ----
#pragma unroll
    for (int i = 0; i < 4; ++i) {
#pragma unroll
        for (int j = 0; j < 2; ++j) {
            const int cg  = colBase + wn * 32 + j * 16 + lane_lo;
            const int rg0 = rowBase + wm * 64 + i * 16 + sel * 8;
            float bval = 0.f;
            if constexpr (MODE & MODE_BIAS) { if (cg < N) bval = bias[cg]; }
#pragma unroll
            for (int e = 0; e < 8; ++e) {
                const int rg = rg0 + e;     // M % 128 == 0 -> rg always valid
                if (cg < N) {
                    float x = acc[i][j][e] + bval;
                    if constexpr (MODE & MODE_LIF) x = (x > 1.0f) ? 1.0f : 0.0f;
                    if constexpr (MODE & MODE_RES) x += res[(size_t)rg * ldr + cg];
                    C[(size_t)rg * ldc + cg] = x;
                }
            }
        }
    }
}

// ---------------------------------------------------------------------------
// Level-0 attention core: 3 tokens, 4 heads, dh=128, temperature 1.
// One block per sample, one wave per head. Writes combined tokens in place
// of q (each wave only touches its own head slice; reads precede writes).
// ---------------------------------------------------------------------------
__global__ __launch_bounds__(128)
void attn3_kernel(const float* __restrict__ q, const float* __restrict__ k,
                  const float* __restrict__ v, float* __restrict__ outp)
{
    const int b    = blockIdx.x;
    const int head = threadIdx.x >> 5;
    const int lane = threadIdx.x & 31;
    const size_t base = (size_t)b * 3 * HDIM + head * 128 + lane * 4;

    float4 qv[3], kv[3], vv[3];
#pragma unroll
    for (int m = 0; m < 3; ++m) {
        qv[m] = *(const float4*)(q + base + m * HDIM);
        kv[m] = *(const float4*)(k + base + m * HDIM);
        vv[m] = *(const float4*)(v + base + m * HDIM);
    }

    const float scale = 0.08838834764831845f;   // 1/sqrt(128)
    float s[3][3];
#pragma unroll
    for (int m = 0; m < 3; ++m)
#pragma unroll
        for (int n = 0; n < 3; ++n) {
            float p = qv[m].x * kv[n].x + qv[m].y * kv[n].y +
                      qv[m].z * kv[n].z + qv[m].w * kv[n].w;
#pragma unroll
            for (int off = 16; off >= 1; off >>= 1)
                p += __shfl_xor(p, off, 32);
            s[m][n] = p * scale;
        }

    float a[3][3];
#pragma unroll
    for (int m = 0; m < 3; ++m) {
        float mx = fmaxf(s[m][0], fmaxf(s[m][1], s[m][2]));
        float e0 = __expf(s[m][0] - mx);
        float e1 = __expf(s[m][1] - mx);
        float e2 = __expf(s[m][2] - mx);
        float inv = 1.0f / (e0 + e1 + e2);
        a[m][0] = e0 * inv; a[m][1] = e1 * inv; a[m][2] = e2 * inv;
    }

#pragma unroll
    for (int m = 0; m < 3; ++m) {
        float4 o;
        o.x = a[m][0] * vv[0].x + a[m][1] * vv[1].x + a[m][2] * vv[2].x;
        o.y = a[m][0] * vv[0].y + a[m][1] * vv[1].y + a[m][2] * vv[2].y;
        o.z = a[m][0] * vv[0].z + a[m][1] * vv[1].z + a[m][2] * vv[2].z;
        o.w = a[m][0] * vv[0].w + a[m][1] * vv[1].w + a[m][2] * vv[2].w;
        *(float4*)(outp + base + m * HDIM) = o;
    }
}

// fused[b,h] = mean over 3 tokens of x[b,m,h]
__global__ __launch_bounds__(256)
void mean3_kernel(const float* __restrict__ x, float* __restrict__ o, int total)
{
    int i = blockIdx.x * blockDim.x + threadIdx.x;
    if (i >= total) return;
    int b = i >> 9;          // /512
    int h = i & 511;
    size_t r = (size_t)b * 3 * HDIM + h;
    o[i] = (x[r] + x[r + HDIM] + x[r + 2 * HDIM]) * (1.0f / 3.0f);
}

// ---------------------------------------------------------------------------
static void gemm(hipStream_t s, const float* A, int lda, const float* W, int ldb,
                 const float* bias, const float* res, int ldr,
                 float* C, int ldc, int M, int N, int K, int mode)
{
    dim3 g((N + 127) / 128, (M + 127) / 128), b(256);
    switch (mode) {
    case 0:
        gemm_wmma<0><<<g, b, 0, s>>>(A, lda, W, ldb, bias, res, ldr, C, ldc, M, N, K); break;
    case MODE_BIAS:
        gemm_wmma<MODE_BIAS><<<g, b, 0, s>>>(A, lda, W, ldb, bias, res, ldr, C, ldc, M, N, K); break;
    case MODE_BIAS | MODE_LIF:
        gemm_wmma<MODE_BIAS | MODE_LIF><<<g, b, 0, s>>>(A, lda, W, ldb, bias, res, ldr, C, ldc, M, N, K); break;
    default:
        gemm_wmma<MODE_BIAS | MODE_LIF | MODE_RES><<<g, b, 0, s>>>(A, lda, W, ldb, bias, res, ldr, C, ldc, M, N, K); break;
    }
}

extern "C" void kernel_launch(void* const* d_in, const int* in_sizes, int n_in,
                              void* d_out, int out_size, void* d_ws, size_t ws_size,
                              hipStream_t stream)
{
    const float* audio  = (const float*)d_in[0];
    const float* vision = (const float*)d_in[1];
    const float* imu    = (const float*)d_in[2];
    const int B  = in_sizes[2] / 256;       // 4096
    const int B3 = B * 3;
    const int H  = HDIM;

    const float* P[64];
    for (int i = 3; i < n_in && i < 64; ++i) P[i] = (const float*)d_in[i];

    // workspace layout (f32)
    float* ws   = (float*)d_ws;
    const size_t szT = (size_t)B * 3 * H;   // token-shaped [B,3,H]
    const size_t szV = (size_t)B * H;       // vector-shaped [B,H]
    float* buf0 = ws;                 // cur / v
    float* buf1 = buf0 + szT;         // tokens / tokens@Wo
    float* buf2 = buf1 + szT;         // q / attn-out
    float* buf3 = buf2 + szT;         // k
    float* buf5 = buf3 + szT;         // fused
    float* buf6 = buf5 + szV;         // level output (residual result)
    float* buf7 = buf6 + szV;         // temp t
    float* buf8 = buf7 + szV;         // temp t@Wv
    (void)ws_size;

    // ---- input projections -> level-0 inputs, written as token slots ----
    gemm(stream, audio,  8192, P[7],  H, P[8],  nullptr, 0, buf0 + 0 * H, 3 * H, B, H, 8192, MODE_BIAS);
    gemm(stream, vision, 3072, P[11], H, P[12], nullptr, 0, buf0 + 1 * H, 3 * H, B, H, 3072, MODE_BIAS);
    gemm(stream, imu,     256, P[9],  H, P[10], nullptr, 0, buf0 + 2 * H, 3 * H, B, H,  256, MODE_BIAS);

    // ---- level 0: per-modality proc + LIF -> tokens [B,3,H] ----
    gemm(stream, buf0 + 0 * H, 3 * H, P[19], H, P[20], nullptr, 0, buf1 + 0 * H, 3 * H, B, H, H, MODE_BIAS | MODE_LIF);
    gemm(stream, buf0 + 1 * H, 3 * H, P[23], H, P[24], nullptr, 0, buf1 + 1 * H, 3 * H, B, H, H, MODE_BIAS | MODE_LIF);
    gemm(stream, buf0 + 2 * H, 3 * H, P[21], H, P[22], nullptr, 0, buf1 + 2 * H, 3 * H, B, H, H, MODE_BIAS | MODE_LIF);
    // q,k,v over all tokens at once (M = 3B); Wq=P[15], Wk=P[13], Wv=P[16]
    gemm(stream, buf1, H, P[15], H, nullptr, nullptr, 0, buf2, H, B3, H, H, 0);
    gemm(stream, buf1, H, P[13], H, nullptr, nullptr, 0, buf3, H, B3, H, H, 0);
    gemm(stream, buf1, H, P[16], H, nullptr, nullptr, 0, buf0, H, B3, H, H, 0);
    attn3_kernel<<<B, 128, 0, stream>>>(buf2, buf3, buf0, buf2);
    gemm(stream, buf2, H, P[14], H, nullptr, nullptr, 0, buf1, H, B3, H, H, 0);   // @Wo
    mean3_kernel<<<(B * H + 255) / 256, 256, 0, stream>>>(buf1, buf5, B * H);
    // out = lif(fused@Wout + b) + fused
    gemm(stream, buf5, H, P[17], H, P[18], buf5, H, buf6, H, B, H, H, MODE_BIAS | MODE_LIF | MODE_RES);

    // ---- levels 1 & 2: single token => attention == (t@Wv)@Wo exactly ----
    for (int l = 1; l <= 2; ++l) {
        const int base = 13 + l * 12;       // 25 / 37
        const float* Wo = P[base + 1];
        const float* Wv = P[base + 3];
        const float* Wou = P[base + 4]; const float* bou = P[base + 5];
        const float* Wp  = P[base + 6]; const float* bp  = P[base + 7];
        gemm(stream, buf6, H, Wp, H, bp, nullptr, 0, buf7, H, B, H, H, MODE_BIAS | MODE_LIF);
        gemm(stream, buf7, H, Wv, H, nullptr, nullptr, 0, buf8, H, B, H, H, 0);
        gemm(stream, buf8, H, Wo, H, nullptr, nullptr, 0, buf5, H, B, H, H, 0);
        gemm(stream, buf5, H, Wou, H, bou, buf5, H, buf6, H, B, H, H, MODE_BIAS | MODE_LIF | MODE_RES);
    }

    // ---- global attention (1 token): g = (out@Wv)@Wo ; then readout ----
    gemm(stream, buf6, H, P[6], H, nullptr, nullptr, 0, buf7, H, B, H, H, 0);
    gemm(stream, buf7, H, P[4], H, nullptr, nullptr, 0, buf8, H, B, H, H, 0);
    gemm(stream, buf8, H, P[41], 1000, P[42], nullptr, 0, (float*)d_out, 1000, B, 1000, H, MODE_BIAS);
}
